// MedianGCN_82368882803059
// MI455X (gfx1250) — matrix operand order repaired and checked by reading, not compiled, not run on hardware
//
#include <hip/hip_runtime.h>
#include <hip/hip_bf16.h>

#define N_NODES 50000
#define N_EDGES 800000
#define NFEAT   256
#define NHID    128
#define NCLASS  32
#define MAX_DEG 48
#define KBUF    25   // k = deg/2 <= 24 -> keep 25 smallest

typedef __bf16 bf16;
typedef __attribute__((ext_vector_type(16))) __bf16 v16bf;
typedef __attribute__((ext_vector_type(8)))  float  v8f;

// ---------------- conversion / setup kernels ----------------

__global__ void k_f32_to_bf16(const float* __restrict__ in, bf16* __restrict__ out, int n) {
    int i = blockIdx.x * blockDim.x + threadIdx.x;
    if (i < n) out[i] = (bf16)in[i];
}

// Pack weight W[K x N] (fp32, row major) into bf16 WMMA B-fragment order:
// Bp[((kt*tilesN + tn)*32 + lane)*16 + e]  holds  W[(kt*32 + 8*(lane>>4) + (e&7) + 16*(e>>3)) * N + tn*16 + (lane&15)]
template <int K, int N>
__global__ void k_pack_w_bf16(const float* __restrict__ W, bf16* __restrict__ Bp) {
    constexpr int tilesN = N / 16;
    int o = blockIdx.x * blockDim.x + threadIdx.x;
    if (o >= K * N) return;
    int e    = o & 15;
    int lane = (o >> 4) & 31;
    int t    = o >> 9;                 // kt*tilesN + tn
    int kt   = t / tilesN;
    int tn   = t - kt * tilesN;
    int kk   = kt * 32 + 8 * (lane >> 4) + (e & 7) + 16 * (e >> 3);
    int n    = tn * 16 + (lane & 15);
    Bp[o] = (bf16)W[kk * N + n];
}

__global__ void k_zero_i32(int* __restrict__ p, int n) {
    int i = blockIdx.x * blockDim.x + threadIdx.x;
    if (i < n) p[i] = 0;
}

// Bucket non-self edges by destination, fixed stride MAX_DEG.
__global__ void k_build_bucket(const int* __restrict__ src, const int* __restrict__ dst,
                               int* __restrict__ cnt, int* __restrict__ bucket) {
    int e = blockIdx.x * blockDim.x + threadIdx.x;
    if (e >= N_EDGES) return;
    int s = src[e], d = dst[e];
    if (s == d) return;                 // remove_self_loops
    int slot = atomicAdd(&cnt[d], 1);
    if (slot < MAX_DEG) bucket[d * MAX_DEG + slot] = s;
}

// ---------------- WMMA bf16 GEMM: C[M,N] = A[M,K] * Bp ----------------
// One wave per 16x16 output tile; K fully unrolled; all fragment loads b128.

template <int K, int N>
__launch_bounds__(256)
__global__ void k_wmma_gemm_bf16(const bf16* __restrict__ A, const bf16* __restrict__ Bp,
                                 float* __restrict__ C, int M) {
    constexpr int tilesN = N / 16;
    int wave = (int)((blockIdx.x * blockDim.x + threadIdx.x) >> 5);
    int lane = threadIdx.x & 31;
    int totalTiles = (M >> 4) * tilesN;
    if (wave >= totalTiles) return;     // wave-uniform: EXEC all-ones at WMMA
    int tm = wave / tilesN;
    int tn = wave - tm * tilesN;
    int half = lane >> 4;
    int l    = lane & 15;

    // A fragment (16-bit A 16x32): M = lane&15; e<8 -> K=k0+e+8*half, e>=8 -> +16
    const bf16* arow = A + (size_t)(tm * 16 + l) * K + 8 * half;
    // Pre-swizzled B: 16 contiguous bf16 per lane per K-tile
    const bf16* brow = Bp + ((size_t)tn * 32 + lane) * 16;
    constexpr size_t bstep = (size_t)tilesN * 32 * 16;   // elements per kt

    v8f acc = {};
#pragma unroll
    for (int kt = 0; kt < K / 32; ++kt) {
        union { v16bf v; uint4 q[2]; } a, b;
        a.q[0] = *(const uint4*)(arow + kt * 32);
        a.q[1] = *(const uint4*)(arow + kt * 32 + 16);
        b.q[0] = *(const uint4*)(brow + kt * bstep);
        b.q[1] = *(const uint4*)(brow + kt * bstep + 8);
        // 8 args: (neg_a, A, neg_b, B, c_mod, C, reuse_a, reuse_b)
        acc = __builtin_amdgcn_wmma_f32_16x16x32_bf16(false, a.v, false, b.v,
                                                      (short)0, acc, false, false);
    }
    // C/D layout: VGPR r -> row = r + 8*half, col = lane&15
#pragma unroll
    for (int r = 0; r < 8; ++r) {
        C[(size_t)(tm * 16 + r + 8 * half) * N + tn * 16 + l] = acc[r];
    }
}

// ---------------- per-destination lower-median aggregation ----------------
// One block per node, one thread per feature; KBUF-smallest kept in a sorted
// register buffer with fully static indexing (no scratch spills).

template <int F, bool RELU, bool OUTBF>
__global__ void k_median(const float* __restrict__ H, const int* __restrict__ cnt,
                         const int* __restrict__ bucket, const float* __restrict__ bias,
                         float* __restrict__ outf, bf16* __restrict__ outb) {
    __shared__ int s_src[MAX_DEG];
    __shared__ int s_deg;
    int node = blockIdx.x;
    int f    = threadIdx.x;

    if (f == 0) {
        int c = cnt[node];
        s_deg = c < MAX_DEG ? c : MAX_DEG;
    }
    __syncthreads();
    int deg = s_deg;
    for (int i = f; i < deg; i += F) s_src[i] = bucket[node * MAX_DEG + i];
    __syncthreads();

    float buf[KBUF];
#pragma unroll
    for (int i = 0; i < KBUF; ++i) buf[i] = __builtin_inff();

    // self-loop message (add_self_loops)
    {
        float x = H[(size_t)node * F + f];
#pragma unroll
        for (int i = 0; i < KBUF; ++i) {
            float lo = fminf(buf[i], x);
            x        = fmaxf(buf[i], x);
            buf[i]   = lo;
        }
    }
    for (int e = 0; e < deg; ++e) {
        float x = H[(size_t)s_src[e] * F + f];
#pragma unroll
        for (int i = 0; i < KBUF; ++i) {
            float lo = fminf(buf[i], x);
            x        = fmaxf(buf[i], x);
            buf[i]   = lo;
        }
    }

    int k = deg >> 1;                   // lower median of deg+1 values
    float med = buf[0];
#pragma unroll
    for (int i = 1; i < KBUF; ++i) med = (i == k) ? buf[i] : med;

    float r = med + bias[f];
    if (RELU) r = fmaxf(r, 0.0f);
    if (OUTBF) outb[(size_t)node * F + f] = (bf16)r;
    else       outf[(size_t)node * F + f] = r;
}

// ---------------- in-place row log_softmax (32 classes == one wave32) ----------------

__global__ void k_log_softmax(float* __restrict__ out) {
    int lane = threadIdx.x & 31;
    int node = blockIdx.x * (blockDim.x >> 5) + (threadIdx.x >> 5);
    if (node >= N_NODES) return;
    float x = out[(size_t)node * NCLASS + lane];
    float m = x;
#pragma unroll
    for (int off = 16; off > 0; off >>= 1) m = fmaxf(m, __shfl_xor(m, off, 32));
    float e = __expf(x - m);
    float s = e;
#pragma unroll
    for (int off = 16; off > 0; off >>= 1) s += __shfl_xor(s, off, 32);
    out[(size_t)node * NCLASS + lane] = (x - m) - __logf(s);
}

// ---------------- launcher ----------------

extern "C" void kernel_launch(void* const* d_in, const int* in_sizes, int n_in,
                              void* d_out, int out_size, void* d_ws, size_t ws_size,
                              hipStream_t stream) {
    const float* features = (const float*)d_in[0];
    const int*   edge     = (const int*)d_in[1];   // [2, N_EDGES] int32
    const float* W1       = (const float*)d_in[2];
    const float* b1       = (const float*)d_in[3];
    const float* W2       = (const float*)d_in[4];
    const float* b2       = (const float*)d_in[5];
    const int* src = edge;
    const int* dst = edge + N_EDGES;

    char* ws = (char*)d_ws;
    size_t off = 0;
    auto take = [&](size_t bytes) -> void* {
        void* p = ws + off;
        off = (off + bytes + 255) & ~(size_t)255;
        return p;
    };
    bf16*  Xb     = (bf16*)take((size_t)N_NODES * NFEAT * sizeof(bf16));
    bf16*  W1p    = (bf16*)take((size_t)NFEAT * NHID * sizeof(bf16));
    bf16*  W2p    = (bf16*)take((size_t)NHID * NCLASS * sizeof(bf16));
    float* H1     = (float*)take((size_t)N_NODES * NHID * sizeof(float));
    bf16*  X2b    = (bf16*)take((size_t)N_NODES * NHID * sizeof(bf16));
    float* H2     = (float*)take((size_t)N_NODES * NCLASS * sizeof(float));
    int*   cnt    = (int*)take((size_t)N_NODES * sizeof(int));
    int*   bucket = (int*)take((size_t)N_NODES * MAX_DEG * sizeof(int));
    (void)ws_size; (void)in_sizes; (void)n_in; (void)out_size;

    const int T = 256;
    k_f32_to_bf16<<<(N_NODES * NFEAT + T - 1) / T, T, 0, stream>>>(features, Xb, N_NODES * NFEAT);
    k_pack_w_bf16<NFEAT, NHID><<<(NFEAT * NHID + T - 1) / T, T, 0, stream>>>(W1, W1p);
    k_pack_w_bf16<NHID, NCLASS><<<(NHID * NCLASS + T - 1) / T, T, 0, stream>>>(W2, W2p);
    k_zero_i32<<<(N_NODES + T - 1) / T, T, 0, stream>>>(cnt, N_NODES);
    k_build_bucket<<<(N_EDGES + T - 1) / T, T, 0, stream>>>(src, dst, cnt, bucket);

    // GEMM1: [50000x256] x [256x128] -> H1   (3125*8 = 25000 waves)
    {
        int tiles  = (N_NODES / 16) * (NHID / 16);
        int blocks = (tiles * 32 + T - 1) / T;
        k_wmma_gemm_bf16<NFEAT, NHID><<<blocks, T, 0, stream>>>(Xb, W1p, H1, N_NODES);
    }
    // median + bias + ReLU -> bf16 X2
    k_median<NHID, true, true><<<N_NODES, NHID, 0, stream>>>(H1, cnt, bucket, b1, nullptr, X2b);

    // GEMM2: [50000x128] x [128x32] -> H2   (3125*2 = 6250 waves)
    {
        int tiles  = (N_NODES / 16) * (NCLASS / 16);
        int blocks = (tiles * 32 + T - 1) / T;
        k_wmma_gemm_bf16<NHID, NCLASS><<<blocks, T, 0, stream>>>(X2b, W2p, H2, N_NODES);
    }
    // median + bias -> d_out (fp32)
    k_median<NCLASS, false, false><<<N_NODES, NCLASS, 0, stream>>>(H2, cnt, bucket, b2,
                                                                   (float*)d_out, nullptr);
    // log_softmax in place
    k_log_softmax<<<(N_NODES + 7) / 8, T, 0, stream>>>((float*)d_out);
}
